// ParallelAttentionBlock_84370337563122
// MI455X (gfx1250) — matrix operand order repaired and checked by reading, not compiled, not run on hardware
//
#include <hip/hip_runtime.h>
#include <cstdint>
#include <cmath>

// ---------------------------------------------------------------- types
typedef _Float16 half_t;
typedef __attribute__((ext_vector_type(16))) _Float16 v16h;
typedef __attribute__((ext_vector_type(8)))  float    v8f;

#define WMMA_F16(a, b, c) \
  __builtin_amdgcn_wmma_f32_16x16x32_f16(false, (a), false, (b), (short)0, (c), false, false)

// CDNA5 async global->LDS tile staging (ASYNCcnt-tracked, no VGPR round trip).
// VDST = LDS byte address (low 32 bits of generic shared pointer), VADDR = 64-bit global.
#define ASYNC_LOAD_B128(ldsoff, gaddr) \
  asm volatile("global_load_async_to_lds_b128 %0, %1, off" :: "v"(ldsoff), "v"(gaddr) : "memory")
#define WAIT_ASYNC0() asm volatile("s_wait_asynccnt 0" ::: "memory")

// ---------------------------------------------------------------- problem dims
constexpr int Bb    = 2;
constexpr int Ss    = 2048;
constexpr int Hh    = 16;
constexpr int Dd    = 64;
constexpr int FFf   = 4096;
constexpr int Ee    = Hh * Dd;           // 1024
constexpr int NPROJ = 3 * Ee + FFf;      // 7168
constexpr int Mrows = Bb * Ss;           // 4096

// ---------------------------------------------------------------- wave-local LDS sync
__device__ __forceinline__ void wave_lds_sync() {
  asm volatile("s_wait_dscnt 0" ::: "memory");
}

// ---------------------------------------------------------------- fragment loaders
// A-matrix 16x32 f16 (MxK), row-major source with leading dim ld (elements).
__device__ __forceinline__ v16h load_a_frag(const half_t* base, int ld, int lane) {
  union { v16h v; uint32_t u[8]; } r;
  const int m  = lane & 15;
  const int hf = lane >> 4;
#pragma unroll
  for (int vi = 0; vi < 8; ++vi) {
    const int k = (vi < 4) ? (2 * vi + 8 * hf) : (16 + 2 * (vi - 4) + 8 * hf);
    r.u[vi] = *(const uint32_t*)(base + (size_t)m * ld + k);
  }
  return r.v;
}

// B-matrix 32x16 f16 (KxN), source stored transposed: element (k,n) at base[n*ld + k].
__device__ __forceinline__ v16h load_bt_frag(const half_t* base, int ld, int lane) {
  union { v16h v; uint32_t u[8]; } r;
  const int n  = lane & 15;
  const int hf = lane >> 4;
#pragma unroll
  for (int vi = 0; vi < 8; ++vi) {
    const int k = (vi < 4) ? (2 * vi + 8 * hf) : (16 + 2 * (vi - 4) + 8 * hf);
    r.u[vi] = *(const uint32_t*)(base + (size_t)n * ld + k);
  }
  return r.v;
}

// ---------------------------------------------------------------- block reduction
__device__ __forceinline__ float block_reduce_sum(float val, float* red) {
  const int t = threadIdx.x;
  red[t] = val;
  __syncthreads();
#pragma unroll
  for (int s = 128; s > 0; s >>= 1) {
    if (t < s) red[t] += red[t + s];
    __syncthreads();
  }
  const float r = red[0];
  __syncthreads();
  return r;
}

// ---------------------------------------------------------------- elementwise cast
__global__ void __launch_bounds__(256) cast_f32_f16_kernel(const float* __restrict__ s,
                                                           half_t* __restrict__ d, int n) {
  int i = blockIdx.x * blockDim.x + threadIdx.x;
  const int st = gridDim.x * blockDim.x;
  for (; i < n; i += st) d[i] = (half_t)s[i];
}

// ---------------------------------------------------------------- V transpose: vt[b][h][d][s]
__global__ void __launch_bounds__(256) vtrans_kernel(const half_t* __restrict__ proj,
                                                     half_t* __restrict__ vt) {
  const size_t total = (size_t)Bb * Hh * Dd * Ss;
  size_t i = (size_t)blockIdx.x * blockDim.x + threadIdx.x;
  const size_t st = (size_t)gridDim.x * blockDim.x;
  for (; i < total; i += st) {
    const int s = (int)(i % Ss);
    size_t r = i / Ss;
    const int d = (int)(r % Dd); r /= Dd;
    const int h = (int)(r % Hh);
    const int b = (int)(r / Hh);
    vt[i] = proj[((size_t)b * Ss + s) * NPROJ + 2 * Ee + h * Dd + d];
  }
}

// ---------------------------------------------------------------- input LayerNorm -> f16
__global__ void __launch_bounds__(256) ln_cast_kernel(const float* __restrict__ x,
                                                      const float* __restrict__ g,
                                                      const float* __restrict__ b,
                                                      half_t* __restrict__ out) {
  __shared__ float red[256];
  const int row = blockIdx.x;
  const int t   = threadIdx.x;
  const float* xr = x + (size_t)row * Ee;
  float v[4];
#pragma unroll
  for (int i = 0; i < 4; ++i) v[i] = xr[t + i * 256];
  const float mean = block_reduce_sum(v[0] + v[1] + v[2] + v[3], red) * (1.0f / Ee);
  float q = 0.f;
#pragma unroll
  for (int i = 0; i < 4; ++i) { const float d = v[i] - mean; q += d * d; }
  const float var = block_reduce_sum(q, red) * (1.0f / Ee);
  const float rs  = rsqrtf(var + 1e-5f);
#pragma unroll
  for (int i = 0; i < 4; ++i) {
    const int c = t + i * 256;
    out[(size_t)row * Ee + c] = (half_t)((v[i] - mean) * rs * g[c] + b[c]);
  }
}

// ---------------------------------------------------------------- QK-LN + GELU (in place on proj f16)
__device__ void ln1024_inplace(half_t* p, const float* g, const float* b, float* red) {
  const int t = threadIdx.x;
  float x[4];
#pragma unroll
  for (int i = 0; i < 4; ++i) x[i] = (float)p[t + i * 256];
  const float mean = block_reduce_sum(x[0] + x[1] + x[2] + x[3], red) * (1.0f / Ee);
  float q = 0.f;
#pragma unroll
  for (int i = 0; i < 4; ++i) { const float d = x[i] - mean; q += d * d; }
  const float var = block_reduce_sum(q, red) * (1.0f / Ee);
  const float rs  = rsqrtf(var + 1e-5f);
#pragma unroll
  for (int i = 0; i < 4; ++i) {
    const int c = t + i * 256;
    p[c] = (half_t)((x[i] - mean) * rs * g[c] + b[c]);
  }
}

__global__ void __launch_bounds__(256) qkln_gelu_kernel(half_t* __restrict__ proj,
                                                        const float* __restrict__ qg,
                                                        const float* __restrict__ qb,
                                                        const float* __restrict__ kg,
                                                        const float* __restrict__ kb) {
  __shared__ float red[256];
  half_t* pr = proj + (size_t)blockIdx.x * NPROJ;
  ln1024_inplace(pr, qg, qb, red);        // q  (cols [0,1024))
  ln1024_inplace(pr + Ee, kg, kb, red);   // k  (cols [1024,2048))
  const int t = threadIdx.x;
#pragma unroll
  for (int j = 0; j < 16; ++j) {
    const int col = 3 * Ee + t + j * 256;
    const float x = (float)pr[col];
    pr[col] = (half_t)(0.5f * x * (1.0f + erff(x * 0.70710678118f)));
  }
}

// ---------------------------------------------------------------- NT GEMM: C = A[M,K] * B[N,K]^T (+addsrc)
// Block tile 128x64, 8 waves (4x2), each wave 32x32 = 2x2 WMMA tiles, K step 32.
// Tiles staged global -> LDS with CDNA5 async loads (ASYNCcnt), no VGPR round trip.
template <typename OutT>
__global__ void __launch_bounds__(256) gemm_nt_kernel(const half_t* __restrict__ A, int lda,
                                                      const half_t* __restrict__ Bm, int ldb,
                                                      OutT* C, int ldc,
                                                      const float* addsrc, int K) {
  __shared__ __attribute__((aligned(16))) half_t Alds[128 * 32];
  __shared__ __attribute__((aligned(16))) half_t Blds[64 * 32];

  const int t  = threadIdx.x;
  const int l  = t & 31;
  const int w  = t >> 5;
  const int wm = w & 3;
  const int wn = w >> 2;

  const half_t* Ablk = A + (size_t)(blockIdx.y * 128) * lda;
  const half_t* Bblk = Bm + (size_t)(blockIdx.x * 64) * ldb;

  v8f zero = {};
  v8f acc[2][2];
  acc[0][0] = zero; acc[0][1] = zero; acc[1][0] = zero; acc[1][1] = zero;

  // per-thread chunk coordinates (fixed across K loop)
  const int ar0 = t >> 2,          ac0 = (t & 3) * 8;          // A chunk 0 (rows 0..63)
  const int ar1 = (t + 256) >> 2,  ac1 = ac0;                  // A chunk 1 (rows 64..127)
  const int br  = t >> 2,          bc  = (t & 3) * 8;          // B chunk  (rows 0..63)
  const uint32_t lds_a0 = (uint32_t)(uintptr_t)&Alds[ar0 * 32 + ac0];
  const uint32_t lds_a1 = (uint32_t)(uintptr_t)&Alds[ar1 * 32 + ac1];
  const uint32_t lds_b  = (uint32_t)(uintptr_t)&Blds[br * 32 + bc];

  for (int k0 = 0; k0 < K; k0 += 32) {
    // async-stage A (128x32) and B (64x32) f16 tiles directly into LDS
    ASYNC_LOAD_B128(lds_a0, (uint64_t)(uintptr_t)(Ablk + (size_t)ar0 * lda + k0 + ac0));
    ASYNC_LOAD_B128(lds_a1, (uint64_t)(uintptr_t)(Ablk + (size_t)ar1 * lda + k0 + ac1));
    ASYNC_LOAD_B128(lds_b,  (uint64_t)(uintptr_t)(Bblk + (size_t)br  * ldb + k0 + bc));
    if (k0 + 32 < K) {  // hint next K-slab toward L2 (global_prefetch_b8)
      __builtin_prefetch(Ablk + (size_t)ar0 * lda + k0 + 32, 0, 0);
      __builtin_prefetch(Bblk + (size_t)br  * ldb + k0 + 32, 0, 0);
    }
    WAIT_ASYNC0();
    __syncthreads();

    const v16h a0 = load_a_frag(&Alds[(wm * 32) * 32], 32, l);
    const v16h a1 = load_a_frag(&Alds[(wm * 32 + 16) * 32], 32, l);
    const v16h b0 = load_bt_frag(&Blds[(wn * 32) * 32], 32, l);
    const v16h b1 = load_bt_frag(&Blds[(wn * 32 + 16) * 32], 32, l);

    acc[0][0] = WMMA_F16(a0, b0, acc[0][0]);
    acc[0][1] = WMMA_F16(a0, b1, acc[0][1]);
    acc[1][0] = WMMA_F16(a1, b0, acc[1][0]);
    acc[1][1] = WMMA_F16(a1, b1, acc[1][1]);
    __syncthreads();
  }

  const int rowbase = blockIdx.y * 128 + wm * 32 + 8 * (l >> 4);
  const int colbase = blockIdx.x * 64 + wn * 32 + (l & 15);
#pragma unroll
  for (int mi = 0; mi < 2; ++mi)
#pragma unroll
    for (int ni = 0; ni < 2; ++ni)
#pragma unroll
      for (int i = 0; i < 8; ++i) {
        const size_t r   = (size_t)(rowbase + mi * 16 + i);
        const size_t c   = (size_t)(colbase + ni * 16);
        const size_t idx = r * (size_t)ldc + c;
        float v = acc[mi][ni][i];
        if (addsrc) v += addsrc[idx];
        C[idx] = (OutT)v;
      }
}

// ---------------------------------------------------------------- causal flash attention
// Grid: B*H*(S/128). Block: 8 waves; wave owns 16 q rows, streams KV in 32-wide tiles.
__global__ void __launch_bounds__(256) attn_kernel(const half_t* __restrict__ proj,
                                                   const half_t* __restrict__ vt,
                                                   half_t* __restrict__ o) {
  __shared__ float  s_s[8][512];
  __shared__ half_t s_p[8][512];
  __shared__ float  s_c[8][16];

  const int t  = threadIdx.x;
  const int l  = t & 31;
  const int w  = t >> 5;
  const int hf = l >> 4;
  const int c0 = l & 15;

  const int qt = blockIdx.x % (Ss / 128);
  const int bh = blockIdx.x / (Ss / 128);
  const int b  = bh / Hh;
  const int h  = bh % Hh;
  const int qbase = qt * 128 + w * 16;

  const half_t* qptr = proj + (size_t)b * Ss * NPROJ + h * Dd;  // q cols [0,E)
  const half_t* kptr = qptr + Ee;                               // k cols [E,2E)
  const half_t* vth  = vt + ((size_t)b * Hh + h) * Dd * Ss;     // vt[h] : [D][S]

  const v16h qa0 = load_a_frag(qptr + (size_t)qbase * NPROJ, NPROJ, l);
  const v16h qa1 = load_a_frag(qptr + (size_t)qbase * NPROJ + 32, NPROJ, l);

  v8f o0 = {}, o1 = {}, o2 = {}, o3 = {};
  float m_run = -1e30f, l_run = 0.f;

  for (int kv0 = 0; kv0 < qbase + 16; kv0 += 32) {
    // ---- scores S = (Q * scale) K^T : 16x32 tile, K-dim 64
    v8f sA = {}, sB = {};
    const v16h kb00 = load_bt_frag(kptr + (size_t)kv0 * NPROJ, NPROJ, l);
    const v16h kb01 = load_bt_frag(kptr + (size_t)kv0 * NPROJ + 32, NPROJ, l);
    const v16h kb10 = load_bt_frag(kptr + (size_t)(kv0 + 16) * NPROJ, NPROJ, l);
    const v16h kb11 = load_bt_frag(kptr + (size_t)(kv0 + 16) * NPROJ + 32, NPROJ, l);
    sA = WMMA_F16(qa0, kb00, sA);
    sA = WMMA_F16(qa1, kb01, sA);
    sB = WMMA_F16(qa0, kb10, sB);
    sB = WMMA_F16(qa1, kb11, sB);

    // ---- scale + causal mask, spill to per-wave LDS
#pragma unroll
    for (int i = 0; i < 8; ++i) {
      const int r = i + 8 * hf;
      float v0 = sA[i] * 0.125f;
      float v1 = sB[i] * 0.125f;
      if (kv0 + c0 > qbase + r) v0 = -1e30f;
      if (kv0 + 16 + c0 > qbase + r) v1 = -1e30f;
      s_s[w][r * 32 + c0]      = v0;
      s_s[w][r * 32 + 16 + c0] = v1;
    }
    wave_lds_sync();

    // ---- online softmax: lane r < 16 owns q-row r
    if (l < 16) {
      float rowm = -1e30f;
#pragma unroll
      for (int j = 0; j < 32; ++j) rowm = fmaxf(rowm, s_s[w][l * 32 + j]);
      const float m_new = fmaxf(m_run, rowm);
      const float corr  = __expf(m_run - m_new);
      float sum = 0.f;
#pragma unroll
      for (int j = 0; j < 32; ++j) {
        const float p = __expf(s_s[w][l * 32 + j] - m_new);
        sum += p;
        s_p[w][l * 32 + j] = (half_t)p;
      }
      l_run = l_run * corr + sum;
      m_run = m_new;
      s_c[w][l] = corr;
    }
    wave_lds_sync();

    // ---- rescale running O
    float cf[8];
#pragma unroll
    for (int i = 0; i < 8; ++i) cf[i] = s_c[w][i + 8 * hf];
#pragma unroll
    for (int i = 0; i < 8; ++i) {
      o0[i] *= cf[i]; o1[i] *= cf[i]; o2[i] *= cf[i]; o3[i] *= cf[i];
    }

    // ---- O += P[16x32] * V[32x64] ; V fragments from pre-transposed vt (k-contiguous)
    const v16h pa  = load_a_frag(&s_p[w][0], 32, l);
    const v16h vb0 = load_bt_frag(vth + (size_t)(0)  * Ss + kv0, Ss, l);
    const v16h vb1 = load_bt_frag(vth + (size_t)(16) * Ss + kv0, Ss, l);
    const v16h vb2 = load_bt_frag(vth + (size_t)(32) * Ss + kv0, Ss, l);
    const v16h vb3 = load_bt_frag(vth + (size_t)(48) * Ss + kv0, Ss, l);
    o0 = WMMA_F16(pa, vb0, o0);
    o1 = WMMA_F16(pa, vb1, o1);
    o2 = WMMA_F16(pa, vb2, o2);
    o3 = WMMA_F16(pa, vb3, o3);
  }

  // ---- finalize
  if (l < 16) s_c[w][l] = 1.0f / l_run;
  wave_lds_sync();
  float cf[8];
#pragma unroll
  for (int i = 0; i < 8; ++i) cf[i] = s_c[w][i + 8 * hf];

  half_t* optr = o + (size_t)b * Ss * Ee + h * Dd;
#pragma unroll
  for (int i = 0; i < 8; ++i) {
    const size_t r = (size_t)(qbase + i + 8 * hf);
    optr[r * Ee + 0  + c0] = (half_t)(o0[i] * cf[i]);
    optr[r * Ee + 16 + c0] = (half_t)(o1[i] * cf[i]);
    optr[r * Ee + 32 + c0] = (half_t)(o2[i] * cf[i]);
    optr[r * Ee + 48 + c0] = (half_t)(o3[i] * cf[i]);
  }
}

// ---------------------------------------------------------------- host launcher
extern "C" void kernel_launch(void* const* d_in, const int* in_sizes, int n_in,
                              void* d_out, int out_size, void* d_ws, size_t ws_size,
                              hipStream_t stream) {
  (void)in_sizes; (void)n_in; (void)out_size; (void)ws_size;
  const float* x      = (const float*)d_in[0];
  const float* w_fus  = (const float*)d_in[1];
  const float* w_attn = (const float*)d_in[2];
  const float* w_ff   = (const float*)d_in[3];
  const float* ln_g   = (const float*)d_in[4];
  const float* ln_b   = (const float*)d_in[5];
  const float* lnq_g  = (const float*)d_in[6];
  const float* lnq_b  = (const float*)d_in[7];
  const float* lnk_g  = (const float*)d_in[8];
  const float* lnk_b  = (const float*)d_in[9];
  float* out = (float*)d_out;

  const size_t N_XH   = (size_t)Mrows * Ee;
  const size_t N_WFH  = (size_t)NPROJ * Ee;
  const size_t N_WAH  = (size_t)Ee * Ee;
  const size_t N_WFFH = (size_t)Ee * FFf;
  const size_t N_PROJ = (size_t)Mrows * NPROJ;
  const size_t N_OH   = (size_t)Mrows * Ee;

  half_t* p    = (half_t*)d_ws;
  half_t* xh   = p;            p += N_XH;
  half_t* wfh  = p;            p += N_WFH;
  half_t* wah  = p;            p += N_WAH;
  half_t* wffh = p;            p += N_WFFH;
  half_t* proj = p;            p += N_PROJ;
  half_t* oh   = p;            p += N_OH;
  half_t* vt   = p;            // [B,H,D,S]

  cast_f32_f16_kernel<<<2048, 256, 0, stream>>>(w_fus, wfh, (int)N_WFH);
  cast_f32_f16_kernel<<<2048, 256, 0, stream>>>(w_attn, wah, (int)N_WAH);
  cast_f32_f16_kernel<<<2048, 256, 0, stream>>>(w_ff, wffh, (int)N_WFFH);

  ln_cast_kernel<<<Mrows, 256, 0, stream>>>(x, ln_g, ln_b, xh);

  gemm_nt_kernel<half_t><<<dim3(NPROJ / 64, Mrows / 128), 256, 0, stream>>>(
      xh, Ee, wfh, Ee, proj, NPROJ, nullptr, Ee);

  qkln_gelu_kernel<<<Mrows, 256, 0, stream>>>(proj, lnq_g, lnq_b, lnk_g, lnk_b);
  vtrans_kernel<<<2048, 256, 0, stream>>>(proj, vt);

  attn_kernel<<<Bb * Hh * (Ss / 128), 256, 0, stream>>>(proj, vt, oh);

  gemm_nt_kernel<float><<<dim3(Ee / 64, Mrows / 128), 256, 0, stream>>>(
      oh, Ee, wah, Ee, out, Ee, x, Ee);

  gemm_nt_kernel<float><<<dim3(Ee / 64, Mrows / 128), 256, 0, stream>>>(
      proj + 3 * Ee, NPROJ, wffh, FFf, out, Ee, out, FFf);
}
